// RoPEAttention_61847529062901
// MI455X (gfx1250) — matrix operand order repaired and checked by reading, not compiled, not run on hardware
//
#include <hip/hip_runtime.h>

// ---------------------------------------------------------------------------
// RoPE attention for MI455X (gfx1250, wave32, WMMA).
// S=4096, D=2048, fp32 in/out; all GEMMs via v_wmma_f32_16x16x32_bf16 with
// double-buffered GLOBAL_LOAD_ASYNC_TO_LDS_B128 staging (ASYNCcnt pipeline).
// ---------------------------------------------------------------------------

typedef __bf16 bf16;
typedef __attribute__((ext_vector_type(16))) __bf16 v16bf;
typedef __attribute__((ext_vector_type(8)))  __bf16 v8bf;
typedef __attribute__((ext_vector_type(4)))  __bf16 v4bf;
typedef __attribute__((ext_vector_type(8)))  float  v8f;

#define S_DIM 4096
#define D_DIM 2048
#define LDSS  40   // 32 bf16 + 8 pad (16B-aligned rows, conflict-free frag loads)

// Async global->LDS copy, 16B per lane. vdst = LDS byte address (VGPR),
// vaddr = 64-bit global address. Tracked by ASYNCcnt.
__device__ __forceinline__ void async_g2l_b128(unsigned lds, const void* g) {
  asm volatile("global_load_async_to_lds_b128 %0, %1, off"
               :: "v"(lds), "v"(g) : "memory");
}
__device__ __forceinline__ void wait_async0() {
  asm volatile("s_wait_asynccnt 0x0" ::: "memory");
}
__device__ __forceinline__ unsigned lds_addr(const void* p) {
  return (unsigned)(unsigned long long)p;  // low 32 bits = LDS offset
}

// ---------------- fp32 -> bf16 conversion (vectorized) ----------------------
__global__ __launch_bounds__(256) void cvt_f32_bf16(
    const float* __restrict__ in, bf16* __restrict__ out, int n4) {
  int i = blockIdx.x * blockDim.x + threadIdx.x;
  if (i >= n4) return;
  float4 v = ((const float4*)in)[i];
  v4bf o = { (bf16)v.x, (bf16)v.y, (bf16)v.z, (bf16)v.w };
  ((v4bf*)out)[i] = o;
}

// ---------------- tiled WMMA GEMM:  C = scale * (A @ B^T) -------------------
// A: [M,K] bf16 row-major, B: [N,K] bf16 row-major (rows read along K).
// mode 0: Cf[M,N] fp32.   mode 1: Ct[N,ldt] bf16 transposed (Ct[n*ldt+m]).
// Block: 256 thr (8 waves), tile 128x128, K-step 32.  Wave tile 32x64.
// Double-buffered async-to-LDS staging overlapped with WMMA compute.
__global__ __launch_bounds__(256) void wmma_gemm(
    const bf16* __restrict__ A, const bf16* __restrict__ B,
    float* __restrict__ Cf, bf16* __restrict__ Ct,
    int M, int N, int K, float scale, int mode, int ldt) {
  __shared__ bf16 lsA[2][128 * LDSS];
  __shared__ bf16 lsB[2][128 * LDSS];

  const int t    = threadIdx.x;
  const int lane = t & 31;
  const int wid  = t >> 5;
  const int wm   = wid & 3;        // 4 waves along M (4*32 = 128)
  const int wn   = wid >> 2;       // 2 waves along N (2*64 = 128)
  const int m0   = blockIdx.y * 128;
  const int n0   = blockIdx.x * 128;
  const int half = (lane >> 4) & 1;
  const int lr   = lane & 15;

  v8f acc[2][4] = {};

  const int row  = t >> 2;         // 64 rows covered by chunk0, +64 by chunk1
  const int col8 = (t & 3) * 8;    // 8 bf16 (16B) per chunk

  // per-thread global sources (advance by 32 bf16 per k-step)
  const bf16* gA0 = A + (size_t)(m0 + row) * K + col8;
  const bf16* gA1 = A + (size_t)(m0 + row + 64) * K + col8;
  const bf16* gB0 = B + (size_t)(n0 + row) * K + col8;
  const bf16* gB1 = B + (size_t)(n0 + row + 64) * K + col8;

  // per-thread LDS destinations, both buffers
  unsigned dA0[2], dA1[2], dB0[2], dB1[2];
#pragma unroll
  for (int b = 0; b < 2; b++) {
    dA0[b] = lds_addr(&lsA[b][row * LDSS + col8]);
    dA1[b] = lds_addr(&lsA[b][(row + 64) * LDSS + col8]);
    dB0[b] = lds_addr(&lsB[b][row * LDSS + col8]);
    dB1[b] = lds_addr(&lsB[b][(row + 64) * LDSS + col8]);
  }

  const int nIter = K >> 5;  // K/32

  // prologue: stage tile 0 into buffer 0
  async_g2l_b128(dA0[0], gA0);
  async_g2l_b128(dA1[0], gA1);
  async_g2l_b128(dB0[0], gB0);
  async_g2l_b128(dB1[0], gB1);

  for (int it = 0; it < nIter; ++it) {
    wait_async0();      // this wave's current-tile async copies have landed
    __syncthreads();    // publish all waves' LDS writes (single barrier/iter)

    if (it + 1 < nIter) {  // stage next tile into the other buffer (overlaps)
      const int b  = (it + 1) & 1;
      const int ko = (it + 1) * 32;
      async_g2l_b128(dA0[b], gA0 + ko);
      async_g2l_b128(dA1[b], gA1 + ko);
      async_g2l_b128(dB0[b], gB0 + ko);
      async_g2l_b128(dB1[b], gB1 + ko);
    }

    const bf16* bufA = lsA[it & 1];
    const bf16* bufB = lsB[it & 1];

    // A fragments: 16x32 bf16; lanes 0-15 hold K 0..7 & 16..23, lanes 16-31
    // hold K 8..15 & 24..31 (ISA 7.12.2 16-bit A layout).
    v16bf afr[2];
#pragma unroll
    for (int mi = 0; mi < 2; mi++) {
      const bf16* pa = &bufA[(wm * 32 + mi * 16 + lr) * LDSS + half * 8];
      v8bf lo = *(const v8bf*)pa;
      v8bf hi = *(const v8bf*)(pa + 16);
      afr[mi] = __builtin_shufflevector(lo, hi, 0, 1, 2, 3, 4, 5, 6, 7,
                                        8, 9, 10, 11, 12, 13, 14, 15);
    }
    // B fragments: 32x16 bf16; lanes 0-15 hold K 0..7 & 16..23, lanes 16-31 K 8..15 & 24..31.
#pragma unroll
    for (int ni = 0; ni < 4; ni++) {
      const bf16* pb = &bufB[(wn * 64 + ni * 16 + lr) * LDSS + half * 8];
      v8bf lo = *(const v8bf*)pb;
      v8bf hi = *(const v8bf*)(pb + 16);
      v16bf bfr = __builtin_shufflevector(lo, hi, 0, 1, 2, 3, 4, 5, 6, 7,
                                          8, 9, 10, 11, 12, 13, 14, 15);
#pragma unroll
      for (int mi = 0; mi < 2; mi++) {
        acc[mi][ni] = __builtin_amdgcn_wmma_f32_16x16x32_bf16(
            false, afr[mi], false, bfr, (short)0, acc[mi][ni], false, false);
      }
    }
  }

  // Epilogue. C/D layout: lane%16 = N col; VGPR r = M row r (lanes<16) or 8+r.
#pragma unroll
  for (int mi = 0; mi < 2; mi++) {
#pragma unroll
    for (int ni = 0; ni < 4; ni++) {
#pragma unroll
      for (int r = 0; r < 8; r++) {
        const int mrow = m0 + wm * 32 + mi * 16 + half * 8 + r;
        const int ncol = n0 + wn * 64 + ni * 16 + lr;
        const float v = acc[mi][ni][r] * scale;
        if (mode == 0) Cf[(size_t)mrow * N + ncol] = v;
        else           Ct[(size_t)ncol * ldt + mrow] = (bf16)v;
      }
    }
  }
}

// ---------------- RoPE: rotate fp32 Q/K, emit bf16 --------------------------
__global__ __launch_bounds__(256) void rope_kernel(
    const float* __restrict__ Qf, const float* __restrict__ Kf,
    bf16* __restrict__ Qr, bf16* __restrict__ Kr) {
  const int s = blockIdx.x;
  const float fs = (float)s;
  const float c0 = -13.287712379549449f / 1024.0f;  // -log2(10000)/half
  for (int i = threadIdx.x; i < 1024; i += 256) {
    float freq = exp2f(c0 * (float)i);
    float sn, cs;
    __sincosf(fs * freq, &sn, &cs);
    size_t idx = (size_t)s * D_DIM + 2 * i;
    float qr = Qf[idx], qi = Qf[idx + 1];
    Qr[idx]     = (bf16)(qr * cs - qi * sn);
    Qr[idx + 1] = (bf16)(qr * sn + qi * cs);
    float kr = Kf[idx], ki = Kf[idx + 1];
    Kr[idx]     = (bf16)(kr * cs - ki * sn);
    Kr[idx + 1] = (bf16)(kr * sn + ki * cs);
  }
}

// ---------------- row softmax in-place (fp32) + bf16 copy -------------------
__global__ __launch_bounds__(256) void softmax_row(
    float* __restrict__ W, bf16* __restrict__ P) {
  const int row = blockIdx.x;
  const int t = threadIdx.x;
  float* w = W + (size_t)row * S_DIM;
  __shared__ float red[8];

  float4 v[4];
  float m = -3.4e38f;
#pragma unroll
  for (int j = 0; j < 4; j++) {
    v[j] = ((const float4*)w)[t + j * 256];
    m = fmaxf(m, fmaxf(fmaxf(v[j].x, v[j].y), fmaxf(v[j].z, v[j].w)));
  }
  for (int off = 16; off; off >>= 1) m = fmaxf(m, __shfl_xor(m, off, 32));
  if ((t & 31) == 0) red[t >> 5] = m;
  __syncthreads();
  float bm = red[0];
#pragma unroll
  for (int i = 1; i < 8; i++) bm = fmaxf(bm, red[i]);

  float e[16];
  float sum = 0.f;
#pragma unroll
  for (int j = 0; j < 4; j++) {
    e[4 * j + 0] = __expf(v[j].x - bm);
    e[4 * j + 1] = __expf(v[j].y - bm);
    e[4 * j + 2] = __expf(v[j].z - bm);
    e[4 * j + 3] = __expf(v[j].w - bm);
    sum += e[4 * j] + e[4 * j + 1] + e[4 * j + 2] + e[4 * j + 3];
  }
  for (int off = 16; off; off >>= 1) sum += __shfl_xor(sum, off, 32);
  __syncthreads();
  if ((t & 31) == 0) red[t >> 5] = sum;
  __syncthreads();
  float bs = 0.f;
#pragma unroll
  for (int i = 0; i < 8; i++) bs += red[i];
  const float inv = 1.0f / bs;

#pragma unroll
  for (int j = 0; j < 4; j++) {
    float4 o = { e[4 * j] * inv, e[4 * j + 1] * inv,
                 e[4 * j + 2] * inv, e[4 * j + 3] * inv };
    ((float4*)w)[t + j * 256] = o;
    v4bf ob = { (bf16)o.x, (bf16)o.y, (bf16)o.z, (bf16)o.w };
    ((v4bf*)(P + (size_t)row * S_DIM))[t + j * 256] = ob;
  }
}

// ---------------------------------------------------------------------------
extern "C" void kernel_launch(void* const* d_in, const int* in_sizes, int n_in,
                              void* d_out, int out_size, void* d_ws,
                              size_t ws_size, hipStream_t stream) {
  const int S = S_DIM, D = D_DIM;
  const size_t SD = (size_t)S * D;   // 8388608
  const size_t DD = (size_t)D * D;   // 4194304
  const size_t SS = (size_t)S * S;   // 16777216

  const float* x  = (const float*)d_in[0];
  const float* Wq = (const float*)d_in[1];
  const float* Wk = (const float*)d_in[2];
  const float* Wv = (const float*)d_in[3];
  float* out     = (float*)d_out;        // [S, D]
  float* weights = out + SD;             // [S, S]

  // workspace carve-up (~193 MB total)
  char* w = (char*)d_ws;
  bf16*  xb  = (bf16*)w;  w += SD * 2;   // x as bf16
  bf16*  wqb = (bf16*)w;  w += DD * 2;
  bf16*  wkb = (bf16*)w;  w += DD * 2;
  bf16*  wvb = (bf16*)w;  w += DD * 2;
  float* Qf  = (float*)w; w += SD * 4;   // pre-RoPE Q fp32
  float* Kf  = (float*)w; w += SD * 4;   // pre-RoPE K fp32
  bf16*  Qrb = (bf16*)w;  w += SD * 2;   // Q_rot bf16 [S,D]
  bf16*  Krb = (bf16*)w;  w += SD * 2;   // K_rot bf16 [S,D]
  bf16*  Vtb = (bf16*)w;  w += SD * 2;   // V^T  bf16 [D,S]
  bf16*  Pb  = (bf16*)w;  w += SS * 2;   // probs bf16 [S,S]

  // 1) convert inputs to bf16
  cvt_f32_bf16<<<(int)(SD / 4 + 255) / 256, 256, 0, stream>>>(x, xb, (int)(SD / 4));
  cvt_f32_bf16<<<(int)(DD / 4 + 255) / 256, 256, 0, stream>>>(Wq, wqb, (int)(DD / 4));
  cvt_f32_bf16<<<(int)(DD / 4 + 255) / 256, 256, 0, stream>>>(Wk, wkb, (int)(DD / 4));
  cvt_f32_bf16<<<(int)(DD / 4 + 255) / 256, 256, 0, stream>>>(Wv, wvb, (int)(DD / 4));

  // 2) projections: Q,K fp32; V stored bf16 transposed [D,S]
  dim3 gQK(D / 128, S / 128);
  wmma_gemm<<<gQK, 256, 0, stream>>>(xb, wqb, Qf, nullptr, S, D, D, 1.0f, 0, 0);
  wmma_gemm<<<gQK, 256, 0, stream>>>(xb, wkb, Kf, nullptr, S, D, D, 1.0f, 0, 0);
  wmma_gemm<<<gQK, 256, 0, stream>>>(xb, wvb, nullptr, Vtb, S, D, D, 1.0f, 1, S);

  // 3) RoPE -> bf16 Q_rot, K_rot
  rope_kernel<<<S, 256, 0, stream>>>(Qf, Kf, Qrb, Krb);

  // 4) scores = (Q_rot @ K_rot^T) / sqrt(D)  -> weights region (fp32)
  dim3 gSc(S / 128, S / 128);
  wmma_gemm<<<gSc, 256, 0, stream>>>(Qrb, Krb, weights, nullptr, S, S, D,
                                     0.022097086912079608f, 0, 0);

  // 5) softmax rows in place; bf16 copy for the PV GEMM
  softmax_row<<<S, 256, 0, stream>>>(weights, Pb);

  // 6) output = probs @ V  (= probs @ Vt^T) -> out region (fp32)
  dim3 gO(D / 128, S / 128);
  wmma_gemm<<<gO, 256, 0, stream>>>(Pb, Vtb, out, nullptr, S, D, S, 1.0f, 0, 0);
}
